// MDTA_75436805587647
// MI455X (gfx1250) — matrix operand order repaired and checked by reading, not compiled, not run on hardware
//
#include <hip/hip_runtime.h>
#include <hip/hip_bf16.h>
#include <math.h>

typedef __attribute__((ext_vector_type(16))) _Float16 v16h;
typedef __attribute__((ext_vector_type(8)))  float    v8f;

#define WMMA_F16(a, b, c) \
  __builtin_amdgcn_wmma_f32_16x16x32_f16(false, (a), false, (b), (short)0, (c), false, false)

constexpr int kB     = 2;
constexpr int kC     = 192;
constexpr int kC3    = 576;
constexpr int kHeads = 4;
constexpr int kHD    = 48;
constexpr int kHDP   = 64;     // padded head dim for K-loop (2 x 32)
constexpr int kNP    = 4096;   // pixels per batch (64x64)

// ---------------------------------------------------------------------------
// Kernel 1a: x [b][c][p] f32 -> xT [b][p][c] f16  (c contiguous)
// ---------------------------------------------------------------------------
__global__ void convert_x_kernel(const float* __restrict__ x, _Float16* __restrict__ xT) {
  int idx = blockIdx.x * blockDim.x + threadIdx.x;  // over b*p
  if (idx >= kB * kNP) return;
  int p = idx & (kNP - 1);
  int b = idx >> 12;
  const float* src = x + ((size_t)b * kC) * kNP + p;
  _Float16* dst = xT + ((size_t)b * kNP + p) * kC;
  for (int c = 0; c < kC; ++c) dst[c] = (_Float16)src[(size_t)c * kNP];
}

// ---------------------------------------------------------------------------
// Kernel 1b: weights f32 -> f16 (row-major kept: [o][c], c contiguous)
// ---------------------------------------------------------------------------
__global__ void convert_w_kernel(const float* __restrict__ w_qkv, const float* __restrict__ w_proj,
                                 _Float16* __restrict__ wq, _Float16* __restrict__ wp) {
  int i = blockIdx.x * blockDim.x + threadIdx.x;
  if (i < kC3 * kC) wq[i] = (_Float16)w_qkv[i];
  int j = i - kC3 * kC;
  if (j >= 0 && j < kC * kC) wp[j] = (_Float16)w_proj[j];
}

// ---------------------------------------------------------------------------
// Generic WMMA GEMM:  out[b][o][p] = sum_c A[o][c] * Bt[b][p][c]  + bias[o]
// A: [M][K] f16 row-major; Bt: [batch][NPIX][K] f16 (K contiguous)
// grid = (NPIX/128, M/16, batch), block = 32 (one wave; 8 N-tiles per wave)
// ---------------------------------------------------------------------------
__global__ void gemm16_kernel(const _Float16* __restrict__ A, const _Float16* __restrict__ Bt,
                              const float* __restrict__ bias, float* __restrict__ out,
                              int M, int K) {
  const int lane = threadIdx.x;
  const int row  = lane & 15;
  const int hi   = lane >> 4;
  const int ng   = blockIdx.x;   // group of 8 N tiles
  const int mt   = blockIdx.y;
  const int bz   = blockIdx.z;

  v8f acc[8] = {};
  const int ksteps = K >> 5;
  for (int kt = 0; kt < ksteps; ++kt) {
    const int k0 = kt << 5;
    // A fragment (16x32): row = lane&15; K = hi*16 + i (contiguous)
    v16h a;
    {
      const _Float16* ap = A + (size_t)(mt * 16 + row) * K + k0 + hi * 16;
#pragma unroll
      for (int i = 0; i < 8; ++i) a[i] = ap[i];
#pragma unroll
      for (int i = 0; i < 8; ++i) a[8 + i] = ap[8 + i];
    }
#pragma unroll
    for (int t = 0; t < 8; ++t) {
      const int p = (ng * 8 + t) * 16 + row;
      // B fragment (32x16): col = lane&15; K = hi*16 + i (contiguous)
      const _Float16* bp = Bt + ((size_t)bz * kNP + p) * K + k0 + hi * 16;
      v16h bv;
#pragma unroll
      for (int i = 0; i < 16; ++i) bv[i] = bp[i];
      acc[t] = WMMA_F16(a, bv, acc[t]);
    }
  }
  // D layout: col = lane&15 (pixel), row = hi*8 + r (out channel)
#pragma unroll
  for (int t = 0; t < 8; ++t) {
    const int p = (ng * 8 + t) * 16 + row;
#pragma unroll
    for (int r = 0; r < 8; ++r) {
      const int o = mt * 16 + hi * 8 + r;
      out[((size_t)bz * M + o) * kNP + p] = acc[t][r] + bias[o];
    }
  }
}

// ---------------------------------------------------------------------------
// Kernel 3: depthwise 3x3 conv (pad 1) + bias, f32 -> f32
// ---------------------------------------------------------------------------
__global__ void dwconv_kernel(const float* __restrict__ in, const float* __restrict__ wdw,
                              const float* __restrict__ bdw, float* __restrict__ out) {
  int idx = blockIdx.x * blockDim.x + threadIdx.x;  // over b*576*4096
  if (idx >= kB * kC3 * kNP) return;
  int p  = idx & (kNP - 1);
  int ch = (idx >> 12) % kC3;
  int b  = idx / (kC3 * kNP);
  int x0 = p & 63, y0 = p >> 6;
  const float* base = in + ((size_t)b * kC3 + ch) * kNP;
  const float* wf   = wdw + ch * 9;
  float s = bdw[ch];
#pragma unroll
  for (int dy = -1; dy <= 1; ++dy) {
    int y = y0 + dy;
    if (y < 0 || y > 63) continue;
#pragma unroll
    for (int dx = -1; dx <= 1; ++dx) {
      int x = x0 + dx;
      if (x < 0 || x > 63) continue;
      s += wf[(dy + 1) * 3 + (dx + 1)] * base[y * 64 + x];
    }
  }
  out[idx] = s;
}

// ---------------------------------------------------------------------------
// Kernel 4: L2-normalize q,k over head dim, pack f16:
//   qT,kT: [b*h][p][64]  (d contiguous, d=48..63 zero)
//   vA:    [b*h][48][p]  (p contiguous)
// ---------------------------------------------------------------------------
__global__ void norm_pack_kernel(const float* __restrict__ dwq, _Float16* __restrict__ qT,
                                 _Float16* __restrict__ kT, _Float16* __restrict__ vA) {
  int idx = blockIdx.x * blockDim.x + threadIdx.x;  // over b*heads*p
  if (idx >= kB * kHeads * kNP) return;
  int p = idx & (kNP - 1);
  int h = (idx >> 12) & 3;
  int b = idx >> 14;
  size_t bh = (size_t)b * kHeads + h;

  float buf[kHD];
  // q (channels 0..191)
  {
    const float* src = dwq + ((size_t)b * kC3 + h * kHD) * kNP + p;
    float ss = 0.f;
    for (int d = 0; d < kHD; ++d) { float v = src[(size_t)d * kNP]; buf[d] = v; ss += v * v; }
    float sc = 1.f / fmaxf(sqrtf(ss), 1e-12f);
    _Float16* dst = qT + (bh * kNP + p) * kHDP;
    for (int d = 0; d < kHD; ++d) dst[d] = (_Float16)(buf[d] * sc);
    for (int d = kHD; d < kHDP; ++d) dst[d] = (_Float16)0.f;
  }
  // k (channels 192..383)
  {
    const float* src = dwq + ((size_t)b * kC3 + kC + h * kHD) * kNP + p;
    float ss = 0.f;
    for (int d = 0; d < kHD; ++d) { float v = src[(size_t)d * kNP]; buf[d] = v; ss += v * v; }
    float sc = 1.f / fmaxf(sqrtf(ss), 1e-12f);
    _Float16* dst = kT + (bh * kNP + p) * kHDP;
    for (int d = 0; d < kHD; ++d) dst[d] = (_Float16)(buf[d] * sc);
    for (int d = kHD; d < kHDP; ++d) dst[d] = (_Float16)0.f;
  }
  // v (channels 384..575), no normalization
  {
    const float* src = dwq + ((size_t)b * kC3 + 2 * kC + h * kHD) * kNP + p;
    for (int d = 0; d < kHD; ++d)
      vA[(bh * kHD + d) * kNP + p] = (_Float16)src[(size_t)d * kNP];
  }
}

// ---------------------------------------------------------------------------
// Kernel 5: flash attention, one wave per (b, h, 32-query block = 2 tiles)
//   S^T[key, query] = K_frag(A) x Q_frag(B); online softmax; O = V(A) x P(B)
//   Each K/V fragment is reused for both query tiles (2 WMMAs per load).
//   output aoT [b][p][c] f16 (c contiguous, ready as GEMM B operand)
// ---------------------------------------------------------------------------
__global__ void flash_attn_kernel(const _Float16* __restrict__ qT, const _Float16* __restrict__ kT,
                                  const _Float16* __restrict__ vA, const float* __restrict__ temp,
                                  _Float16* __restrict__ aoT) {
  const int lane = threadIdx.x;
  const int col  = lane & 15;
  const int hi   = lane >> 4;
  const int qb   = blockIdx.x;   // 0..127 (32 queries each)
  const int h    = blockIdx.y;   // 0..3
  const int b    = blockIdx.z;   // 0..1
  const size_t bh = (size_t)b * kHeads + h;
  const float tscale = temp[h];

  // Q B-fragments: 2 query tiles x 2 d-steps (loop invariant)
  v16h qf[2][2];
#pragma unroll
  for (int u = 0; u < 2; ++u)
#pragma unroll
    for (int s = 0; s < 2; ++s) {
      const _Float16* qq =
          qT + (bh * kNP + qb * 32 + u * 16 + col) * kHDP + s * 32 + hi * 16;
#pragma unroll
      for (int i = 0; i < 16; ++i) qf[u][s][i] = qq[i];
    }

  v8f   accO[2][3] = {};
  float mrow[2] = {-INFINITY, -INFINITY};
  float lrow[2] = {0.f, 0.f};

  for (int kb = 0; kb < kNP / 32; ++kb) {
    if (kb + 1 < kNP / 32)
      __builtin_prefetch(kT + (bh * kNP + (kb + 1) * 32 + col) * kHDP, 0, 3);

    // ---- S^T tiles: tile t covers keys kb*32 + t*16 .. +15, both q tiles ----
    v8f st[2][2] = {};
#pragma unroll
    for (int t = 0; t < 2; ++t) {
      const _Float16* kp = kT + (bh * kNP + kb * 32 + t * 16 + col) * kHDP;
#pragma unroll
      for (int s = 0; s < 2; ++s) {
        v16h kf;
        const _Float16* kk = kp + s * 32 + hi * 16;
#pragma unroll
        for (int i = 0; i < 8; ++i) kf[i] = kk[i];
#pragma unroll
        for (int i = 0; i < 8; ++i) kf[8 + i] = kk[8 + i];
#pragma unroll
        for (int u = 0; u < 2; ++u) st[u][t] = WMMA_F16(kf, qf[u][s], st[u][t]);
      }
    }

    // ---- online softmax per query tile (query = lane pair col / col+16) ----
    v16h pf[2];
#pragma unroll
    for (int u = 0; u < 2; ++u) {
      float s0[8], s1[8];
      float lmax = -INFINITY;
#pragma unroll
      for (int r = 0; r < 8; ++r) {
        s0[r] = st[u][0][r] * tscale;
        s1[r] = st[u][1][r] * tscale;
        lmax = fmaxf(lmax, fmaxf(s0[r], s1[r]));
      }
      float mblk = fmaxf(lmax, __shfl_xor(lmax, 16, 32));
      float mnew = fmaxf(mrow[u], mblk);
      float alpha = __expf(mrow[u] - mnew);
      float lsum = 0.f;
#pragma unroll
      for (int r = 0; r < 8; ++r) {
        s0[r] = __expf(s0[r] - mnew);
        s1[r] = __expf(s1[r] - mnew);
        lsum += s0[r] + s1[r];
      }
      lrow[u] = lrow[u] * alpha + lsum;
      mrow[u] = mnew;
#pragma unroll
      for (int t = 0; t < 3; ++t)
#pragma unroll
        for (int r = 0; r < 8; ++r) accO[u][t][r] *= alpha;

      // repack P (C-layout f32 tiles) into B-fragment (f16)
      // B frag lane L (query L&15): elements = keys (L<16 ? 0..15 : 16..31)
#pragma unroll
      for (int r = 0; r < 8; ++r) {
        float t0p = __shfl_xor(s0[r], 16, 32);
        float t1p = __shfl_xor(s1[r], 16, 32);
        float lo = (hi == 0) ? s0[r] : t1p;
        float hv = (hi == 0) ? t0p   : s1[r];
        pf[u][r]     = (_Float16)lo;
        pf[u][8 + r] = (_Float16)hv;
      }
    }

    // ---- O += V x P : A = V tile [16 d x 32 keys], reused for both q tiles ----
#pragma unroll
    for (int t = 0; t < 3; ++t) {
      v16h vf;
      const _Float16* vp = vA + (bh * kHD + t * 16 + col) * kNP + kb * 32 + hi * 16;
#pragma unroll
      for (int i = 0; i < 8; ++i) vf[i] = vp[i];
#pragma unroll
      for (int i = 0; i < 8; ++i) vf[8 + i] = vp[8 + i];
#pragma unroll
      for (int u = 0; u < 2; ++u) accO[u][t] = WMMA_F16(vf, pf[u], accO[u][t]);
    }
  }

  // ---- finalize: divide by row sums, store f16 to aoT [b][p][c] ----
#pragma unroll
  for (int u = 0; u < 2; ++u) {
    float ltot = lrow[u] + __shfl_xor(lrow[u], 16, 32);
    float inv = 1.f / ltot;
    const int p = qb * 32 + u * 16 + col;   // D layout: col = query
#pragma unroll
    for (int t = 0; t < 3; ++t)
#pragma unroll
      for (int r = 0; r < 8; ++r) {
        const int c = h * kHD + t * 16 + hi * 8 + r;  // row = hi*8 + r
        aoT[((size_t)b * kNP + p) * kC + c] = (_Float16)(accO[u][t][r] * inv);
      }
  }
}

// ---------------------------------------------------------------------------
// Host-side launcher
// ---------------------------------------------------------------------------
extern "C" void kernel_launch(void* const* d_in, const int* in_sizes, int n_in,
                              void* d_out, int out_size, void* d_ws, size_t ws_size,
                              hipStream_t stream) {
  const float* x      = (const float*)d_in[0];
  const float* w_qkv  = (const float*)d_in[1];
  const float* b_qkv  = (const float*)d_in[2];
  const float* w_dw   = (const float*)d_in[3];
  const float* b_dw   = (const float*)d_in[4];
  const float* w_proj = (const float*)d_in[5];
  const float* b_proj = (const float*)d_in[6];
  const float* temp   = (const float*)d_in[7];
  float* out = (float*)d_out;

  // workspace layout (bytes, 256-aligned)
  char* ws = (char*)d_ws;
  float*    qkv = (float*)(ws);                              // 18,874,368
  float*    dwq = (float*)(ws + 18874368);                   // 18,874,368
  _Float16* xT  = (_Float16*)(ws + 37748736);                //  3,145,728
  _Float16* wq  = (_Float16*)(ws + 40894464);                //    221,184
  _Float16* wp  = (_Float16*)(ws + 41115648);                //     73,728
  _Float16* qTm = (_Float16*)(ws + 41189376);                //  4,194,304
  _Float16* kTm = (_Float16*)(ws + 45383680);                //  4,194,304
  _Float16* vAm = (_Float16*)(ws + 49577984);                //  3,145,728
  _Float16* aoT = (_Float16*)(ws + 52723712);                //  3,145,728  (end 55,869,440)

  // 1. precision / layout conversion
  convert_x_kernel<<<(kB * kNP + 255) / 256, 256, 0, stream>>>(x, xT);
  convert_w_kernel<<<(kC3 * kC + kC * kC + 255) / 256, 256, 0, stream>>>(w_qkv, w_proj, wq, wp);

  // 2. qkv 1x1 conv as WMMA GEMM: [576,192] x [192,4096] per batch
  gemm16_kernel<<<dim3(kNP / 128, kC3 / 16, kB), 32, 0, stream>>>(wq, xT, b_qkv, qkv, kC3, kC);

  // 3. depthwise 3x3 + bias
  dwconv_kernel<<<(kB * kC3 * kNP + 255) / 256, 256, 0, stream>>>(qkv, w_dw, b_dw, dwq);

  // 4. l2-normalize q,k over head dim; pack f16 operands
  norm_pack_kernel<<<(kB * kHeads * kNP + 255) / 256, 256, 0, stream>>>(dwq, qTm, kTm, vAm);

  // 5. flash attention (one wave per 32-query block, 2x K/V fragment reuse)
  flash_attn_kernel<<<dim3(kNP / 32, kHeads, kB), 32, 0, stream>>>(qTm, kTm, vAm, temp, aoT);

  // 6. output projection as WMMA GEMM -> d_out (f32) + bias
  gemm16_kernel<<<dim3(kNP / 128, kC / 16, kB), 32, 0, stream>>>(wp, aoT, b_proj, out, kC, kC);
}
